// GLMBlock_87342454931953
// MI455X (gfx1250) — matrix operand order repaired and checked by reading, not compiled, not run on hardware
//
#include <hip/hip_runtime.h>

// ---------------- Types ----------------
typedef __attribute__((ext_vector_type(16))) __bf16 v16bf;
typedef __attribute__((ext_vector_type(8)))  __bf16 v8bf;
typedef __attribute__((ext_vector_type(8)))  float  v8f;

#define B_ 2
#define S_ 1024
#define H_ 32
#define HKV_ 2
#define D_ 128
#define HID_ 4096
#define FFN_ 16384
#define T_ 2048
#define NQKV_ 4608
#define ALPHA_ 7.48331477354788277116870f   // sqrt(56)
#define SCALE_ 0.08838834764831845f         // 1/sqrt(128)
#define EPS_ 1e-5f

static __device__ __forceinline__ __bf16 f2bf(float f) {
    union { float f; unsigned u; } v; v.f = f;
    unsigned r = v.u + 0x7FFFu + ((v.u >> 16) & 1u);
    unsigned short h = (unsigned short)(r >> 16);
    return __builtin_bit_cast(__bf16, h);
}
static __device__ __forceinline__ int imin(int a, int b) { return a < b ? a : b; }
static __device__ __forceinline__ int imax(int a, int b) { return a > b ? a : b; }

// A-fragment (16x32, lane = row M, elems 0-15 -> K = kh*16 + e)
static __device__ __forceinline__ v16bf load_fragA(const __bf16* base, int stride, int lane) {
    const __bf16* p = base + (size_t)(lane & 15) * stride;
    int kh = lane >> 4;
    v8bf lo = *(const v8bf*)(p + kh * 16);
    v8bf hi = *(const v8bf*)(p + kh * 16 + 8);
    return __builtin_shufflevector(lo, hi, 0,1,2,3,4,5,6,7,8,9,10,11,12,13,14,15);
}
// B-fragment (32x16, lane = col N, elems 0-15 -> K = kh*16 + e). base = row-major (N,K) tile.
static __device__ __forceinline__ v16bf load_fragB(const __bf16* base, int stride, int lane) {
    const __bf16* p = base + (size_t)(lane & 15) * stride + (lane >> 4) * 16;
    v8bf lo = *(const v8bf*)(p);
    v8bf hi = *(const v8bf*)(p + 8);
    return __builtin_shufflevector(lo, hi, 0,1,2,3,4,5,6,7,8,9,10,11,12,13,14,15);
}
static __device__ __forceinline__ v8f wmma_bf16(v16bf a, v16bf b, v8f c) {
    return __builtin_amdgcn_wmma_f32_16x16x32_bf16(false, a, false, b, (short)0, c, false, false);
}

// ---------------- Weight convert + transpose: f32 (K,N) -> bf16 (N,K) ----------------
__global__ __launch_bounds__(256) void w_cvt_t(const float* __restrict__ W,
                                               __bf16* __restrict__ Wt, int K, int N) {
    __shared__ __align__(16) __bf16 tile[32][33];
    int n0 = blockIdx.x * 32, k0 = blockIdx.y * 32;
    int tx = threadIdx.x & 31, ty = threadIdx.x >> 5;
    #pragma unroll
    for (int i = ty; i < 32; i += 8)
        tile[i][tx] = f2bf(W[(size_t)(k0 + i) * N + n0 + tx]);
    __syncthreads();
    #pragma unroll
    for (int i = ty; i < 32; i += 8)
        Wt[(size_t)(n0 + i) * K + k0 + tx] = tile[tx][i];
}

// ---------------- Fused residual-add + LayerNorm (fp32 out + bf16 out) ----------------
__global__ __launch_bounds__(256) void ln_kernel(const float* __restrict__ x,
                                                 const float* __restrict__ skip,
                                                 const float* __restrict__ w,
                                                 const float* __restrict__ bb,
                                                 float alpha,
                                                 float* __restrict__ outf,
                                                 __bf16* __restrict__ outb) {
    __shared__ float red[256];
    int row = blockIdx.x, tid = threadIdx.x;
    const float* xr = x + (size_t)row * HID_;
    const float* sr = skip + (size_t)row * HID_;
    float v[16]; float s = 0.f;
    #pragma unroll
    for (int i = 0; i < 16; i++) {
        int c = tid * 16 + i;
        float t = xr[c] + alpha * sr[c];
        v[i] = t; s += t;
    }
    red[tid] = s; __syncthreads();
    for (int st = 128; st > 0; st >>= 1) { if (tid < st) red[tid] += red[tid + st]; __syncthreads(); }
    float mean = red[0] * (1.f / HID_);
    __syncthreads();
    float vs = 0.f;
    #pragma unroll
    for (int i = 0; i < 16; i++) { v[i] -= mean; vs += v[i] * v[i]; }
    red[tid] = vs; __syncthreads();
    for (int st = 128; st > 0; st >>= 1) { if (tid < st) red[tid] += red[tid + st]; __syncthreads(); }
    float inv = rsqrtf(red[0] * (1.f / HID_) + EPS_);
    #pragma unroll
    for (int i = 0; i < 16; i++) {
        int c = tid * 16 + i;
        float o = v[i] * inv * w[c] + bb[c];
        outf[(size_t)row * HID_ + c] = o;
        outb[(size_t)row * HID_ + c] = f2bf(o);
    }
}

// ---------------- WMMA bf16 GEMM: C(M,N)f32 = A(M,K)bf16 @ Bt(N,K)bf16^T ----------------
// Block: 256 threads = 8 waves (2x4), block tile 128x256, wave tile 64x64.
__global__ __launch_bounds__(256) void gemm_bf16(const __bf16* __restrict__ A,
                                                 const __bf16* __restrict__ Bt,
                                                 float* __restrict__ C,
                                                 int M, int N, int K) {
    int lane = threadIdx.x & 31;
    int wave = threadIdx.x >> 5;
    int bm = blockIdx.y * 128 + (wave >> 2) * 64;
    int bn = blockIdx.x * 256 + (wave & 3) * 64;
    v8f acc[4][4] = {};
    for (int k0 = 0; k0 < K; k0 += 32) {
        v16bf af[4], bf[4];
        #pragma unroll
        for (int i = 0; i < 4; i++) af[i] = load_fragA(A + (size_t)(bm + i * 16) * K + k0, K, lane);
        #pragma unroll
        for (int j = 0; j < 4; j++) bf[j] = load_fragB(Bt + (size_t)(bn + j * 16) * K + k0, K, lane);
        #pragma unroll
        for (int i = 0; i < 4; i++)
            #pragma unroll
            for (int j = 0; j < 4; j++)
                acc[i][j] = wmma_bf16(af[i], bf[j], acc[i][j]);
    }
    int kh = lane >> 4, col = lane & 15;
    #pragma unroll
    for (int i = 0; i < 4; i++)
        #pragma unroll
        for (int j = 0; j < 4; j++)
            #pragma unroll
            for (int r = 0; r < 8; r++) {
                int m = bm + i * 16 + r + 8 * kh;
                int n = bn + j * 16 + col;
                C[(size_t)m * N + n] = acc[i][j][r];
            }
}

// ---------------- RoPE (2-segment) + layout: q(B,H,S,D) k(B,HKV,S,D) vT(B,HKV,D,S), bf16 ----------------
__global__ __launch_bounds__(256) void rope_kernel(const float* __restrict__ qkv,
                                                   const int* __restrict__ start_pos,
                                                   const int* __restrict__ first_seqlen,
                                                   __bf16* __restrict__ qb,
                                                   __bf16* __restrict__ kb,
                                                   __bf16* __restrict__ vtb) {
    int idx = blockIdx.x * blockDim.x + threadIdx.x;   // T*36*64 threads
    int i  = idx & 63;
    int hh = (idx >> 6) % 36;
    int t  = idx / (64 * 36);
    int b = t / S_, s = t % S_;
    const float* src = qkv + ((size_t)t * 36 + hh) * D_;
    int half = i >> 5;          // 0: dims 0..63 (pos1), 1: dims 64..127 (pos2)
    int j = i & 31;
    int d0 = half * 64 + 2 * j;
    float x1 = src[d0], x2 = src[d0 + 1];
    float o1, o2;
    if (hh < H_ + HKV_) {
        int id = start_pos[b] + s;
        int fs = first_seqlen[b];
        int pos = (half == 0) ? imin(id, fs - 1) : imax(id - fs + 1, 0);
        float inv = __expf(-(float)(2 * j) * (1.f / 64.f) * 9.2103403719761836f); // 10000^(-2j/64)
        float ang = (float)pos * inv;
        float sn, cs; __sincosf(ang, &sn, &cs);
        o1 = x1 * cs - x2 * sn;
        o2 = x1 * sn + x2 * cs;
    } else { o1 = x1; o2 = x2; }
    if (hh < H_) {
        size_t base = (((size_t)b * H_ + hh) * S_ + s) * D_;
        qb[base + d0] = f2bf(o1); qb[base + d0 + 1] = f2bf(o2);
    } else if (hh < H_ + HKV_) {
        size_t base = (((size_t)b * HKV_ + (hh - H_)) * S_ + s) * D_;
        kb[base + d0] = f2bf(o1); kb[base + d0 + 1] = f2bf(o2);
    } else {
        size_t base = ((size_t)b * HKV_ + (hh - H_ - HKV_)) * (size_t)D_ * S_;
        vtb[base + (size_t)d0 * S_ + s]       = f2bf(o1);
        vtb[base + (size_t)(d0 + 1) * S_ + s] = f2bf(o2);
    }
}

// ---------------- Flash attention, WMMA bf16. 1 wave per (b,h,16-query tile). ----------------
__global__ __launch_bounds__(128) void attn_kernel(const __bf16* __restrict__ qb,
                                                   const __bf16* __restrict__ kb,
                                                   const __bf16* __restrict__ vtb,
                                                   const float* __restrict__ mask,
                                                   __bf16* __restrict__ attnb) {
    __shared__ __align__(16) __bf16 Pb[4][16][32];
    int lane = threadIdx.x & 31;
    int wv   = threadIdx.x >> 5;
    int gw = blockIdx.x * 4 + wv;
    int qt = gw & 63;                 // S/16 = 64
    int h  = (gw >> 6) & 31;
    int b  = gw >> 11;
    int hk = h >> 4;                  // H/HKV = 16
    int q0 = qt * 16;
    int kh = lane >> 4, ln = lane & 15;

    const __bf16* Qp = qb  + (((size_t)b * H_ + h) * S_ + q0) * D_;
    const __bf16* Kp = kb  + (((size_t)b * HKV_ + hk) * S_) * D_;
    const __bf16* Vt = vtb + ((size_t)b * HKV_ + hk) * (size_t)D_ * S_;
    const float*  Mp = mask + ((size_t)b * S_ + q0) * S_;

    v16bf qf[4];
    #pragma unroll
    for (int i = 0; i < 4; i++) qf[i] = load_fragA(Qp + i * 32, D_, lane);

    v8f o[8] = {};
    float rm[8], rl[8];
    #pragma unroll
    for (int r = 0; r < 8; r++) { rm[r] = -1e30f; rl[r] = 0.f; }

    for (int k0 = 0; k0 < S_; k0 += 32) {
        // scores: two 16x16 tiles over keys [k0,k0+16) and [k0+16,k0+32)
        v8f s0 = {}, s1 = {};
        #pragma unroll
        for (int i = 0; i < 4; i++) {
            v16bf kf0 = load_fragB(Kp + (size_t)k0 * D_ + i * 32,        D_, lane);
            v16bf kf1 = load_fragB(Kp + (size_t)(k0 + 16) * D_ + i * 32, D_, lane);
            s0 = wmma_bf16(qf[i], kf0, s0);
            s1 = wmma_bf16(qf[i], kf1, s1);
        }
        // online softmax (row M = r + 8*kh spans the 16 lanes of this half)
        #pragma unroll
        for (int r = 0; r < 8; r++) {
            int m = r + 8 * kh;
            float a0 = s0[r] * SCALE_ + Mp[(size_t)m * S_ + k0 + ln];
            float a1 = s1[r] * SCALE_ + Mp[(size_t)m * S_ + k0 + 16 + ln];
            float mx = fmaxf(a0, a1);
            mx = fmaxf(mx, __shfl_xor(mx, 1, 32));
            mx = fmaxf(mx, __shfl_xor(mx, 2, 32));
            mx = fmaxf(mx, __shfl_xor(mx, 4, 32));
            mx = fmaxf(mx, __shfl_xor(mx, 8, 32));
            float mnew = fmaxf(rm[r], mx);
            float sc = __expf(rm[r] - mnew);
            float p0 = __expf(a0 - mnew);
            float p1 = __expf(a1 - mnew);
            float sum = p0 + p1;
            sum += __shfl_xor(sum, 1, 32);
            sum += __shfl_xor(sum, 2, 32);
            sum += __shfl_xor(sum, 4, 32);
            sum += __shfl_xor(sum, 8, 32);
            rl[r] = rl[r] * sc + sum;
            rm[r] = mnew;
            #pragma unroll
            for (int t = 0; t < 8; t++) o[t][r] *= sc;
            Pb[wv][m][ln]      = f2bf(p0);
            Pb[wv][m][16 + ln] = f2bf(p1);
        }
        // re-layout P (C-frag) -> A-frag via per-wave LDS (same-wave ordering)
        v16bf pf = load_fragA(&Pb[wv][0][0], 32, lane);
        #pragma unroll
        for (int t = 0; t < 8; t++) {
            v16bf vf = load_fragB(Vt + (size_t)(t * 16) * S_ + k0, S_, lane);
            o[t] = wmma_bf16(pf, vf, o[t]);
        }
    }
    // normalize + emit bf16 in (T, H*D) layout for the Wo GEMM
    #pragma unroll
    for (int t = 0; t < 8; t++)
        #pragma unroll
        for (int r = 0; r < 8; r++) {
            int m = q0 + r + 8 * kh;
            float val = o[t][r] / rl[r];
            attnb[(((size_t)b * S_ + m) * H_ + h) * D_ + t * 16 + ln] = f2bf(val);
        }
}

// ---------------- GELU (tanh approx) f32 -> bf16 ----------------
__global__ __launch_bounds__(256) void gelu_kernel(const float* __restrict__ in,
                                                   __bf16* __restrict__ out, long long n) {
    long long i0 = ((long long)blockIdx.x * blockDim.x + threadIdx.x) * 4;
    #pragma unroll
    for (int i = 0; i < 4; i++) {
        long long c = i0 + i;
        float x = in[c];
        float g = 0.5f * x * (1.f + tanhf(0.7978845608028654f * (x + 0.044715f * x * x * x)));
        out[c] = f2bf(g);
    }
}

// ---------------- Host-side orchestration ----------------
extern "C" void kernel_launch(void* const* d_in, const int* in_sizes, int n_in,
                              void* d_out, int out_size, void* d_ws, size_t ws_size,
                              hipStream_t stream) {
    const float* x      = (const float*)d_in[0];
    const float* skip   = (const float*)d_in[1];
    const float* amask  = (const float*)d_in[2];
    const int*   spos   = (const int*)d_in[6];
    const int*   fseq   = (const int*)d_in[7];
    const float* ln1w   = (const float*)d_in[13];
    const float* ln1b   = (const float*)d_in[14];
    const float* ln2w   = (const float*)d_in[15];
    const float* ln2b   = (const float*)d_in[16];
    const float* wqkv   = (const float*)d_in[17];
    const float* wo     = (const float*)d_in[18];
    const float* w1     = (const float*)d_in[19];
    const float* w2     = (const float*)d_in[20];

    char* ws = (char*)d_ws;
    size_t off = 0;
    auto alloc = [&](size_t bytes) { size_t o = off; off += (bytes + 255) & ~(size_t)255; return o; };

    __bf16* Wqkv = (__bf16*)(ws + alloc((size_t)NQKV_ * HID_ * 2));   // (N,K) transposed
    __bf16* Wo   = (__bf16*)(ws + alloc((size_t)HID_  * HID_ * 2));
    __bf16* W1   = (__bf16*)(ws + alloc((size_t)FFN_  * HID_ * 2));
    __bf16* W2   = (__bf16*)(ws + alloc((size_t)HID_  * FFN_ * 2));
    float*  h1f  = (float*) (ws + alloc((size_t)T_ * HID_ * 4));
    __bf16* h2b  = (__bf16*)(ws + alloc((size_t)T_ * HID_ * 2));
    __bf16* gel  = (__bf16*)(ws + alloc((size_t)T_ * FFN_ * 2));

    // transient region (reused by ffn1f after attention phase is done)
    size_t trans = off;
    __bf16* h1b  = (__bf16*)(ws + alloc((size_t)T_ * HID_ * 2));
    float*  qkvf = (float*) (ws + alloc((size_t)T_ * NQKV_ * 4));
    __bf16* qb   = (__bf16*)(ws + alloc((size_t)B_ * H_ * S_ * D_ * 2));
    __bf16* kb   = (__bf16*)(ws + alloc((size_t)B_ * HKV_ * S_ * D_ * 2));
    __bf16* vtb  = (__bf16*)(ws + alloc((size_t)B_ * HKV_ * D_ * S_ * 2));
    __bf16* attb = (__bf16*)(ws + alloc((size_t)T_ * HID_ * 2));
    float*  aof  = (float*) (ws + alloc((size_t)T_ * HID_ * 4));
    float*  ffn1 = (float*)(ws + trans);   // 134 MB, overlaps dead transients above

    float* out_ffn = (float*)d_out;                    // (T, HID)
    float* out_h2  = (float*)d_out + (size_t)T_ * HID_;

    // 1) weights -> bf16 transposed
    w_cvt_t<<<dim3(NQKV_ / 32, HID_ / 32), 256, 0, stream>>>(wqkv, Wqkv, HID_, NQKV_);
    w_cvt_t<<<dim3(HID_  / 32, HID_ / 32), 256, 0, stream>>>(wo,   Wo,   HID_, HID_);
    w_cvt_t<<<dim3(FFN_  / 32, HID_ / 32), 256, 0, stream>>>(w1,   W1,   HID_, FFN_);
    w_cvt_t<<<dim3(HID_  / 32, FFN_ / 32), 256, 0, stream>>>(w2,   W2,   FFN_, HID_);

    // 2) h1 = LN(x + alpha*skip)
    ln_kernel<<<T_, 256, 0, stream>>>(x, skip, ln1w, ln1b, ALPHA_, h1f, h1b);

    // 3) qkv = h1 @ wqkv
    gemm_bf16<<<dim3(NQKV_ / 256, T_ / 128), 256, 0, stream>>>(h1b, Wqkv, qkvf, T_, NQKV_, HID_);

    // 4) RoPE + split/transpose
    rope_kernel<<<(T_ * 36 * 64) / 256, 256, 0, stream>>>(qkvf, spos, fseq, qb, kb, vtb);

    // 5) flash attention -> bf16 (T, H*D)
    attn_kernel<<<(B_ * H_ * (S_ / 16)) / 4, 128, 0, stream>>>(qb, kb, vtb, amask, attb);

    // 6) attn_out = attn @ wo
    gemm_bf16<<<dim3(HID_ / 256, T_ / 128), 256, 0, stream>>>(attb, Wo, aof, T_, HID_, HID_);

    // 7) h2 = LN(attn_out + alpha*h1)  -> second output + bf16
    ln_kernel<<<T_, 256, 0, stream>>>(aof, h1f, ln2w, ln2b, ALPHA_, out_h2, h2b);

    // 8) ffn1 = h2 @ w1
    gemm_bf16<<<dim3(FFN_ / 256, T_ / 128), 256, 0, stream>>>(h2b, W1, ffn1, T_, FFN_, HID_);

    // 9) gelu
    gelu_kernel<<<(int)(((long long)T_ * FFN_) / 1024), 256, 0, stream>>>(ffn1, gel, (long long)T_ * FFN_);

    // 10) ffn_out = gelu @ w2 -> first output
    gemm_bf16<<<dim3(HID_ / 256, T_ / 128), 256, 0, stream>>>(gel, W2, out_ffn, T_, HID_, FFN_);
}
